// LePEAttention_38938173506000
// MI455X (gfx1250) — matrix operand order repaired and checked by reading, not compiled, not run on hardware
//
#include <hip/hip_runtime.h>

// ---------------- problem constants ----------------
#define RR     12
#define LL     1728      // RR^3
#define CC     768
#define NHEADS 12
#define HD     64        // head dim
#define BATCH  4

#define KBLOCK 64        // keys per LDS stage
#define LDSS   80        // LDS row stride in halves (160B: keeps 16B/32B alignment, staggers banks)

typedef __attribute__((ext_vector_type(16))) _Float16 v16h;
typedef __attribute__((ext_vector_type(8)))  _Float16 v8h;
typedef __attribute__((ext_vector_type(4)))  _Float16 v4h;
typedef __attribute__((ext_vector_type(8)))  float    v8f;
typedef __attribute__((ext_vector_type(4)))  float    v4f;

static __device__ __forceinline__ v8f wmma_f16(v16h a, v16h b, v8f c) {
  // D = A(16x32 f16) * B(32x16 f16) + C(16x16 f32)
  return __builtin_amdgcn_wmma_f32_16x16x32_f16(
      /*neg_a=*/false, a, /*neg_b=*/false, b,
      /*c_mod=*/(short)0, c, /*reuse_a=*/false, /*reuse_b=*/false);
}

static __device__ __forceinline__ v16h cat8(v8h lo, v8h hi) {
  return __builtin_shufflevector(lo, hi, 0,1,2,3,4,5,6,7,8,9,10,11,12,13,14,15);
}

// A-operand fragment of the K tile: 16 keys x 32 dims starting at dbase.
// Per-lane A layout: halves 0..15 -> dim dbase+16*half+i
static __device__ __forceinline__ v16h load_afrag(const _Float16* rowp, int dbase, int half) {
  const _Float16* p = rowp + dbase + 16 * half;
  v8h lo = *(const v8h*)p;
  v8h hi = *(const v8h*)(p + 8);
  return cat8(lo, hi);
}

// ---------------- flash attention kernel ----------------
// grid: BATCH*NHEADS*(LL/64) blocks of 64 threads (2 waves; each wave owns 32 queries
// as two 16-row Q tiles -> every K/V fragment loaded from LDS feeds two WMMAs)
__global__ __launch_bounds__(64) void lepe_attn_kernel(
    const float* __restrict__ qg, const float* __restrict__ kg,
    const float* __restrict__ vg, float* __restrict__ out)
{
  __shared__ __align__(16) _Float16 Kh[KBLOCK][LDSS];  // [key][dim]
  __shared__ __align__(16) _Float16 Vt[HD][LDSS];      // [dim][key] (transposed)

  const int tid  = threadIdx.x;
  const int lane = tid & 31;
  const int wave = tid >> 5;
  const int half = lane >> 4;   // 0/1
  const int l16  = lane & 15;

  const int bid  = blockIdx.x;
  const int qblk = bid % (LL / 64);
  const int h    = (bid / (LL / 64)) % NHEADS;
  const int b    = bid / ((LL / 64) * NHEADS);
  const int q0   = qblk * 64 + wave * 32;      // this wave's 32 queries (2 tiles of 16)

  // ---- load Q fragments (B operand of S^T = K*Q^T), scale folded in ----
  // tile u, frag f: lane holds Q[q0+16u+l16][32*f + 16*half + j], j=0..15
  const float scale = 0.125f; // 1/sqrt(64)
  v16h bQ0[2], bQ1[2];
#pragma unroll
  for (int u = 0; u < 2; ++u) {
    const float* qrow = qg + ((size_t)(b * LL + q0 + 16 * u + l16) * CC) + h * HD;
    const int d0 = 16 * half;
    const int d1 = 32 + 16 * half;
#pragma unroll
    for (int j = 0; j < 16; ++j) {
      bQ0[u][j] = (_Float16)(qrow[d0 + j] * scale);
      bQ1[u][j] = (_Float16)(qrow[d1 + j] * scale);
    }
  }

  // O accumulator per q-tile: 4 dim-tiles of 16x16 f32 (lane = dim column, vgpr = query row)
  v8f O0[2] = {{}, {}}, O1[2] = {{}, {}}, O2[2] = {{}, {}}, O3[2] = {{}, {}};
  float m_i[2] = {-3.0e38f, -3.0e38f};
  float l_i[2] = {0.0f, 0.0f};

  for (int kb = 0; kb < LL; kb += KBLOCK) {
    __syncthreads();  // previous block's LDS reads done
    // ---- cooperative stage: K -> Kh row-major, V -> Vt transposed (f32 -> f16) ----
#pragma unroll
    for (int i = 0; i < 16; ++i) {
      const int flat = i * 64 + tid;         // 1024 float4 slots = 64x64 elems
      const int row  = flat >> 4;            // key within block
      const int c4   = (flat & 15) << 2;     // dim (multiple of 4)
      const size_t gidx = ((size_t)(b * LL + kb + row) * CC) + h * HD + c4;
      v4f kv = *(const v4f*)(kg + gidx);
      v4f vv = *(const v4f*)(vg + gidx);
      if (kb + KBLOCK < LL) {                // warm next block (global_prefetch_b8)
        __builtin_prefetch(kg + gidx + (size_t)KBLOCK * CC, 0, 0);
        __builtin_prefetch(vg + gidx + (size_t)KBLOCK * CC, 0, 0);
      }
      v4h khv;
      khv[0] = (_Float16)kv[0]; khv[1] = (_Float16)kv[1];
      khv[2] = (_Float16)kv[2]; khv[3] = (_Float16)kv[3];
      *(v4h*)&Kh[row][c4] = khv;
      Vt[c4 + 0][row] = (_Float16)vv[0];
      Vt[c4 + 1][row] = (_Float16)vv[1];
      Vt[c4 + 2][row] = (_Float16)vv[2];
      Vt[c4 + 3][row] = (_Float16)vv[3];
    }
    __syncthreads();

#pragma unroll
    for (int step = 0; step < 2; ++step) {
      const int k0 = step * 32;  // 32 keys this inner step

      // ---- GEMM1: S^T tiles (keys x queries); each K a-frag feeds both q-tiles ----
      const _Float16* krow0 = &Kh[k0 + l16][0];        // keys k0..k0+15
      const _Float16* krow1 = &Kh[k0 + 16 + l16][0];   // keys k0+16..k0+31
      v8f st0[2] = {{}, {}}, st1[2] = {{}, {}};
      {
        v16h aK = load_afrag(krow0, 0, half);
        st0[0] = wmma_f16(aK, bQ0[0], st0[0]);
        st0[1] = wmma_f16(aK, bQ0[1], st0[1]);
        aK = load_afrag(krow0, 32, half);
        st0[0] = wmma_f16(aK, bQ1[0], st0[0]);
        st0[1] = wmma_f16(aK, bQ1[1], st0[1]);
        aK = load_afrag(krow1, 0, half);
        st1[0] = wmma_f16(aK, bQ0[0], st1[0]);
        st1[1] = wmma_f16(aK, bQ0[1], st1[1]);
        aK = load_afrag(krow1, 32, half);
        st1[0] = wmma_f16(aK, bQ1[0], st1[0]);
        st1[1] = wmma_f16(aK, bQ1[1], st1[1]);
      }

      // ---- online softmax per q-tile: lane's 16 values all belong to query (q0+16u+l16) ----
      v16h aP[2];
#pragma unroll
      for (int u = 0; u < 2; ++u) {
        float mx = -3.0e38f;
#pragma unroll
        for (int i = 0; i < 8; ++i) { mx = fmaxf(mx, st0[u][i]); mx = fmaxf(mx, st1[u][i]); }
        mx = fmaxf(mx, __shfl_xor(mx, 16, 32));      // combine keys held by lane^16
        const float m_new = fmaxf(m_i[u], mx);

        float p0[8], p1[8], s = 0.0f;
#pragma unroll
        for (int i = 0; i < 8; ++i) {
          p0[i] = __expf(st0[u][i] - m_new);
          p1[i] = __expf(st1[u][i] - m_new);
          s += p0[i] + p1[i];
        }
        s += __shfl_xor(s, 16, 32);
        const float alpha = __expf(m_i[u] - m_new);
        l_i[u] = l_i[u] * alpha + s;
        m_i[u] = m_new;

        // pack P directly into A-operand layout (register-only transpose)
#pragma unroll
        for (int i = 0; i < 8; ++i) {
          aP[u][i]     = (_Float16)p0[i];   // keys k0 + 8*half + i
          aP[u][i + 8] = (_Float16)p1[i];   // keys k0 + 16 + 8*half + i
        }

        // rescale O (per query row = vgpr index; broadcast alpha by shuffle)
#pragma unroll
        for (int v = 0; v < 8; ++v) {
          const float av = __shfl(alpha, v + 8 * half, 32);
          O0[u][v] *= av; O1[u][v] *= av; O2[u][v] *= av; O3[u][v] *= av;
        }
      }

      // ---- GEMM2: O += P * V; each V b-frag feeds both q-tiles ----
#pragma unroll
      for (int t = 0; t < 4; ++t) {
        const _Float16* vp = &Vt[16 * t + l16][k0 + 8 * half];
        v16h bV = cat8(*(const v8h*)vp, *(const v8h*)(vp + 16));
        if (t == 0) { O0[0] = wmma_f16(aP[0], bV, O0[0]); O0[1] = wmma_f16(aP[1], bV, O0[1]); }
        if (t == 1) { O1[0] = wmma_f16(aP[0], bV, O1[0]); O1[1] = wmma_f16(aP[1], bV, O1[1]); }
        if (t == 2) { O2[0] = wmma_f16(aP[0], bV, O2[0]); O2[1] = wmma_f16(aP[1], bV, O2[1]); }
        if (t == 3) { O3[0] = wmma_f16(aP[0], bV, O3[0]); O3[1] = wmma_f16(aP[1], bV, O3[1]); }
      }
    }
  }

  // ---- epilogue: normalize by row-sum and store [B, L, C] ----
#pragma unroll
  for (int u = 0; u < 2; ++u) {
    const float rcp = 1.0f / l_i[u];   // valid for query q0+16u+l16 (both half-lanes hold it)
#pragma unroll
    for (int v = 0; v < 8; ++v) {
      const int m = v + 8 * half;                   // query row held in this vgpr slot
      const float rv = __shfl(rcp, m, 32);
      const size_t base = ((size_t)(b * LL + q0 + 16 * u + m) * CC) + h * HD + l16;
      out[base +  0] = O0[u][v] * rv;
      out[base + 16] = O1[u][v] * rv;
      out[base + 32] = O2[u][v] * rv;
      out[base + 48] = O3[u][v] * rv;
    }
  }
}

// ---------------- depthwise 3x3x3 LePE conv, accumulate into out ----------------
// grid: BATCH*LL*(CC/256) blocks of 256 threads; thread -> one (b,l,c) element
__global__ __launch_bounds__(256) void lepe_add_kernel(
    const float* __restrict__ vg, const float* __restrict__ w,
    const float* __restrict__ bias, float* __restrict__ out)
{
  const int bid  = blockIdx.x;
  const int cblk = bid % (CC / 256);
  const int l    = (bid / (CC / 256)) % LL;
  const int b    = bid / ((CC / 256) * LL);
  const int c    = cblk * 256 + threadIdx.x;

  const int z = l / (RR * RR);
  const int y = (l / RR) % RR;
  const int x = l % RR;

  float acc = bias[c];
  const float* wc = w + c * 27;
#pragma unroll
  for (int dz = -1; dz <= 1; ++dz) {
    const int zz = z + dz;
    if ((unsigned)zz >= (unsigned)RR) continue;
#pragma unroll
    for (int dy = -1; dy <= 1; ++dy) {
      const int yy = y + dy;
      if ((unsigned)yy >= (unsigned)RR) continue;
#pragma unroll
      for (int dx = -1; dx <= 1; ++dx) {
        const int xx = x + dx;
        if ((unsigned)xx >= (unsigned)RR) continue;
        const int lp = zz * (RR * RR) + yy * RR + xx;
        acc += wc[(dz + 1) * 9 + (dy + 1) * 3 + (dx + 1)] *
               vg[((size_t)(b * LL + lp)) * CC + c];
      }
    }
  }
  const size_t o = ((size_t)(b * LL + l)) * CC + c;
  out[o] += acc;
}

extern "C" void kernel_launch(void* const* d_in, const int* in_sizes, int n_in,
                              void* d_out, int out_size, void* d_ws, size_t ws_size,
                              hipStream_t stream) {
  (void)in_sizes; (void)n_in; (void)out_size; (void)d_ws; (void)ws_size;
  const float* qkv    = (const float*)d_in[0];   // [3, B, L, C] fp32
  const float* lepe_w = (const float*)d_in[1];   // [C, 1, 3, 3, 3]
  const float* lepe_b = (const float*)d_in[2];   // [C]
  float* out = (float*)d_out;                    // [B, L, C] fp32

  const size_t plane = (size_t)BATCH * LL * CC;
  const float* qg = qkv;
  const float* kg = qkv + plane;
  const float* vg = qkv + 2 * plane;

  lepe_attn_kernel<<<dim3(BATCH * NHEADS * (LL / 64)), dim3(64), 0, stream>>>(qg, kg, vg, out);
  lepe_add_kernel<<<dim3(BATCH * LL * (CC / 256)), dim3(256), 0, stream>>>(vg, lepe_w, lepe_b, out);
}
